// AlignmentFrequencyAttention_43645457662091
// MI455X (gfx1250) — matrix operand order repaired and checked by reading, not compiled, not run on hardware
//
#include <hip/hip_runtime.h>
#include <hip/hip_bf16.h>

// ---------------------------------------------------------------------------
// AlignmentFrequencyAttention for MI455X (gfx1250, wave32, WMMA)
//
//  splitQK    : q/k -> f16 hi/lo planes pre-permuted into WMMA A-fragment order
//  fillBasis1 : cos/sin DFT basis planes [1040 x 2048] (f16 hi/lo split)
//  pass1      : full-spectrum |DFT| sums per bin via v_wmma_f32_16x16x32_f16
//               (3-term hi/lo split => ~f32 accuracy), zero-repack operands
//  topk       : top-64 bins (k: bin0 zeroed), jax tie-break
//  pass2      : DFT at selected bins -> Xq_sel / Xk_sel (complex)
//  bmm1/2/3   : complex 64^3 GEMMs + complex tanh + W projection (VALU)
//  prepG      : irfft-as-GEMM A operand, scattered into fragment order
//  fillBasis2 : cos/sin synthesis basis planes
//  synth      : out[8192x2048] = G[8192x128] * Basis2 via WMMA (no LDS)
// ---------------------------------------------------------------------------

typedef _Float16 v16h __attribute__((ext_vector_type(16)));
typedef float    v8f  __attribute__((ext_vector_type(8)));
typedef _Float16 h2   __attribute__((ext_vector_type(2)));
typedef _Float16 half_t;

#define NL    2048
#define NBIN  1025
#define NBP   1040
#define NSER  8192
#define NTOPK 64
#define NBH   128

__device__ __forceinline__ h2 splitf(float x) {
    h2 r;
    _Float16 h = (_Float16)x;
    r.x = h;
    r.y = (_Float16)(x - (float)h);
    return r;
}

__device__ __forceinline__ float2 cmulf(float2 a, float2 b) {
    return make_float2(a.x * b.x - a.y * b.y, a.x * b.y + a.y * b.x);
}

__device__ __forceinline__ float2 ctanhf2(float2 z) {
    float a2 = 2.f * z.x, b2 = 2.f * z.y;
    if (fabsf(a2) > 60.f) return make_float2(a2 > 0.f ? 1.f : -1.f, 0.f);
    float ea = expf(a2), em = expf(-a2);
    float d  = 0.5f * (ea + em) + cosf(b2);
    return make_float2(0.5f * (ea - em) / d, sinf(b2) / d);
}

// Fragment-order position of element (M, Klocal) inside a 16x32 f16 A tile:
// lane = (K>>4)*16 + M ; elem = K&15 ; pos = lane*16 + elem
__device__ __forceinline__ int afragPos(int M, int K) {
    return ((K >> 4) * 16 + M) * 16 + (K & 15);
}

// --------------------------------------------------------------------------
__global__ void initBins(float* binsum) {
    int i = blockIdx.x * blockDim.x + threadIdx.x;
    if (i < 2 * NBP) binsum[i] = 0.f;
}

// Split q/k into hi/lo planes, permuted to WMMA A-fragment order.
// grid (512*64 tile-chunks, 1, 2 sel), block 256.
__global__ __launch_bounds__(256) void splitQK(
    const float* __restrict__ q, const float* __restrict__ k,
    half_t* __restrict__ AHq, half_t* __restrict__ ALq,
    half_t* __restrict__ AHk, half_t* __restrict__ ALk)
{
    const int sel = blockIdx.z;
    const float* src = sel ? k : q;
    half_t* AH = sel ? AHk : AHq;
    half_t* AL = sel ? ALk : ALq;

    const int stile = blockIdx.x >> 6, chunk = blockIdx.x & 63;
    const int tid = threadIdx.x;
    const int s0 = stile * 16;
    const float* base = src + (size_t)(s0 >> 9) * (size_t)(NL * 512) + (s0 & 511);

    __shared__ half_t shH[512];
    __shared__ half_t shL[512];
    #pragma unroll
    for (int r = 0; r < 2; ++r) {
        int eidx = tid + r * 256;           // eidx = j*16 + i
        int i = eidx & 15, j = eidx >> 4;
        h2 sp = splitf(base[(size_t)(chunk * 32 + j) * 512 + i]);
        int pos = afragPos(i, j);
        shH[pos] = sp.x;
        shL[pos] = sp.y;
    }
    __syncthreads();
    size_t obase = (size_t)blockIdx.x * 512;
    #pragma unroll
    for (int r = 0; r < 2; ++r) {
        int eidx = tid + r * 256;
        AH[obase + eidx] = shH[eidx];
        AL[obase + eidx] = shL[eidx];
    }
}

// Basis1 planes: [f][t], exact integer phase reduction (f*t) mod 2048.
__global__ void fillBasis1(half_t* __restrict__ CH, half_t* __restrict__ CL,
                           half_t* __restrict__ SH, half_t* __restrict__ SL)
{
    int gid = blockIdx.x * blockDim.x + threadIdx.x;   // 1040*2048 exact
    int f = gid >> 11, t = gid & (NL - 1);
    float c = 0.f, s = 0.f;
    if (f < NBIN) {
        int phase = (f * t) & (NL - 1);
        double th = (double)phase * (3.14159265358979323846 / 1024.0);
        c = (float)cos(th);
        s = (float)sin(th);
    }
    h2 cc = splitf(c), ss = splitf(s);
    CH[gid] = cc.x; CL[gid] = cc.y;
    SH[gid] = ss.x; SL[gid] = ss.y;
}

// --------------------------------------------------------------------------
// Pass 1: per-bin sum over series of |DFT|. All operands direct v16h loads.
// grid (512 series-tiles, 9 bin-groups, 2 q/k), block 256 (8 waves x 16 bins)
__global__ __launch_bounds__(256) void pass1(
    const half_t* __restrict__ AHq, const half_t* __restrict__ ALq,
    const half_t* __restrict__ AHk, const half_t* __restrict__ ALk,
    const half_t* __restrict__ CH, const half_t* __restrict__ CL,
    const half_t* __restrict__ SH, const half_t* __restrict__ SL,
    float* __restrict__ binsum)
{
    const int sel = blockIdx.z;
    const half_t* AH = sel ? AHk : AHq;
    const half_t* AL = sel ? ALk : ALq;
    const int stile = blockIdx.x;
    const int tid = threadIdx.x;
    const int wave = tid >> 5, lane = tid & 31;
    const int lm = lane & 15, khalf = lane >> 4;

    int btRaw = blockIdx.y * 8 + wave;
    const bool valid = (btRaw <= 64);
    const int binbase = (valid ? btRaw : 64) * 16;

    size_t aofs = (size_t)(stile * 64) * 512 + lane * 16;
    size_t bofs = (size_t)(binbase + lm) * NL + khalf * 16;

    v8f accC = {}; v8f accS = {};

    for (int chunk = 0; chunk < 64; ++chunk) {
        v16h ah = *(const v16h*)(AH + aofs);
        v16h al = *(const v16h*)(AL + aofs);
        v16h bch = *(const v16h*)(CH + bofs);
        v16h bcl = *(const v16h*)(CL + bofs);
        v16h bsh = *(const v16h*)(SH + bofs);
        v16h bsl = *(const v16h*)(SL + bofs);
        if (chunk < 63) {
            __builtin_prefetch(AH + aofs + 512, 0, 1);
            __builtin_prefetch(CH + bofs + 32, 0, 1);
            __builtin_prefetch(SH + bofs + 32, 0, 1);
        }
        accC = __builtin_amdgcn_wmma_f32_16x16x32_f16(false, ah, false, bch, (short)0, accC, false, false);
        accC = __builtin_amdgcn_wmma_f32_16x16x32_f16(false, al, false, bch, (short)0, accC, false, false);
        accC = __builtin_amdgcn_wmma_f32_16x16x32_f16(false, ah, false, bcl, (short)0, accC, false, false);
        accS = __builtin_amdgcn_wmma_f32_16x16x32_f16(false, ah, false, bsh, (short)0, accS, false, false);
        accS = __builtin_amdgcn_wmma_f32_16x16x32_f16(false, al, false, bsh, (short)0, accS, false, false);
        accS = __builtin_amdgcn_wmma_f32_16x16x32_f16(false, ah, false, bsl, (short)0, accS, false, false);
        aofs += 512;
        bofs += 32;
    }

    float msum = 0.f;
    #pragma unroll
    for (int r = 0; r < 8; ++r)
        msum += sqrtf(accC[r] * accC[r] + accS[r] * accS[r]);
    msum += __shfl_xor(msum, 16, 32);

    int f = binbase + lm;
    if (valid && lane < 16 && f < NBIN)
        atomicAdd(&binsum[sel * NBP + f], msum);
}

// --------------------------------------------------------------------------
__global__ void topk(const float* __restrict__ binsum, int* __restrict__ idxout) {
    const int sel = blockIdx.x;
    const int tid = threadIdx.x;
    __shared__ float vals[NBP];
    __shared__ float bv[256];
    __shared__ int   bi[256];

    for (int i = tid; i < NBP; i += 256) {
        float v = (i < NBIN) ? binsum[sel * NBP + i] : -1.f;
        if (sel == 1 && i == 0) v = -1.f;
        vals[i] = v;
    }
    __syncthreads();

    for (int it = 0; it < NTOPK; ++it) {
        float best = -2.f; int besti = 0;
        for (int i = tid; i < NBP; i += 256) {
            float v = vals[i];
            if (v > best || (v == best && i < besti)) { best = v; besti = i; }
        }
        bv[tid] = best; bi[tid] = besti;
        __syncthreads();
        for (int off = 128; off > 0; off >>= 1) {
            if (tid < off) {
                float v = bv[tid + off]; int j = bi[tid + off];
                if (v > bv[tid] || (v == bv[tid] && j < bi[tid])) { bv[tid] = v; bi[tid] = j; }
            }
            __syncthreads();
        }
        if (tid == 0) { idxout[sel * NTOPK + it] = bi[0]; vals[bi[0]] = -1.f; }
        __syncthreads();
    }
}

// --------------------------------------------------------------------------
// Pass 2: DFT at selected bins. grid (512,1,2), block 128 (4 waves x 16 modes)
__global__ __launch_bounds__(128) void pass2(
    const half_t* __restrict__ AHq, const half_t* __restrict__ ALq,
    const half_t* __restrict__ AHk, const half_t* __restrict__ ALk,
    const half_t* __restrict__ CH, const half_t* __restrict__ CL,
    const half_t* __restrict__ SH, const half_t* __restrict__ SL,
    const int* __restrict__ idx, float2* __restrict__ Xq, float2* __restrict__ Xk)
{
    const int sel = blockIdx.z;
    const half_t* AH = sel ? AHk : AHq;
    const half_t* AL = sel ? ALk : ALq;
    float2* dst = sel ? Xk : Xq;
    const int stile = blockIdx.x;
    const int tid = threadIdx.x;
    const int wave = tid >> 5, lane = tid & 31;
    const int lm = lane & 15, khalf = lane >> 4;
    const int modebase = wave * 16;

    const int f = idx[sel * NTOPK + modebase + lm];
    size_t aofs = (size_t)(stile * 64) * 512 + lane * 16;
    size_t bofs = (size_t)f * NL + khalf * 16;

    v8f accC = {}; v8f accS = {};

    for (int chunk = 0; chunk < 64; ++chunk) {
        v16h ah = *(const v16h*)(AH + aofs);
        v16h al = *(const v16h*)(AL + aofs);
        v16h bch = *(const v16h*)(CH + bofs);
        v16h bcl = *(const v16h*)(CL + bofs);
        v16h bsh = *(const v16h*)(SH + bofs);
        v16h bsl = *(const v16h*)(SL + bofs);
        accC = __builtin_amdgcn_wmma_f32_16x16x32_f16(false, ah, false, bch, (short)0, accC, false, false);
        accC = __builtin_amdgcn_wmma_f32_16x16x32_f16(false, al, false, bch, (short)0, accC, false, false);
        accC = __builtin_amdgcn_wmma_f32_16x16x32_f16(false, ah, false, bcl, (short)0, accC, false, false);
        accS = __builtin_amdgcn_wmma_f32_16x16x32_f16(false, ah, false, bsh, (short)0, accS, false, false);
        accS = __builtin_amdgcn_wmma_f32_16x16x32_f16(false, al, false, bsh, (short)0, accS, false, false);
        accS = __builtin_amdgcn_wmma_f32_16x16x32_f16(false, ah, false, bsl, (short)0, accS, false, false);
        aofs += 512;
        bofs += 32;
    }

    #pragma unroll
    for (int r = 0; r < 8; ++r) {
        int s = stile * 16 + r + (khalf ? 8 : 0);
        dst[(size_t)s * NTOPK + modebase + lm] = make_float2(accC[r], -accS[r]);
    }
}

// --------------------------------------------------------------------------
// bmm1: xqk[x,y] = sum_e Q[e,x] * K[e,y], then complex tanh. grid 128 (b,h).
__global__ __launch_bounds__(256) void bmm1(
    const float2* __restrict__ Xq, const float2* __restrict__ Xk,
    float2* __restrict__ T)
{
    const int bh = blockIdx.x, tid = threadIdx.x;
    __shared__ float2 Qs[32][64];
    __shared__ float2 Ks[32][64];
    const int x0 = (tid >> 4) * 4, y0 = (tid & 15) * 4;
    float2 acc[4][4] = {};

    for (int ec = 0; ec < 2; ++ec) {
        __syncthreads();
        for (int idx = tid; idx < 2048; idx += 256) {
            int e = idx >> 6, xx = idx & 63;
            Qs[e][xx] = Xq[(size_t)(bh * 64 + ec * 32 + e) * NTOPK + xx];
            Ks[e][xx] = Xk[(size_t)(bh * 64 + ec * 32 + e) * NTOPK + xx];
        }
        __syncthreads();
        for (int e = 0; e < 32; ++e) {
            float2 qv[4], kv[4];
            #pragma unroll
            for (int i = 0; i < 4; ++i) { qv[i] = Qs[e][x0 + i]; kv[i] = Ks[e][y0 + i]; }
            #pragma unroll
            for (int i = 0; i < 4; ++i)
                #pragma unroll
                for (int j = 0; j < 4; ++j) {
                    float2 p = cmulf(qv[i], kv[j]);
                    acc[i][j].x += p.x; acc[i][j].y += p.y;
                }
        }
    }
    #pragma unroll
    for (int i = 0; i < 4; ++i)
        #pragma unroll
        for (int j = 0; j < 4; ++j)
            T[(size_t)bh * 4096 + (x0 + i) * 64 + (y0 + j)] = ctanhf2(acc[i][j]);
}

// bmm2: xqkv[e,x] = sum_y T[x,y] * K[e,y]; stored as [bh][x][e]. grid 128.
__global__ __launch_bounds__(256) void bmm2(
    const float2* __restrict__ T, const float2* __restrict__ Xk,
    float2* __restrict__ xqkv)
{
    const int bh = blockIdx.x, tid = threadIdx.x;
    __shared__ float2 Ts[64][32];
    __shared__ float2 Ks[64][32];
    const int e0 = (tid >> 4) * 4, x0 = (tid & 15) * 4;
    float2 acc[4][4] = {};

    for (int yc = 0; yc < 2; ++yc) {
        __syncthreads();
        for (int idx = tid; idx < 2048; idx += 256) {
            int a = idx >> 5, y = idx & 31;
            Ts[a][y] = T[(size_t)bh * 4096 + a * 64 + yc * 32 + y];
            Ks[a][y] = Xk[(size_t)(bh * 64 + a) * NTOPK + yc * 32 + y];
        }
        __syncthreads();
        for (int y = 0; y < 32; ++y) {
            float2 tv[4], kv[4];
            #pragma unroll
            for (int i = 0; i < 4; ++i) { tv[i] = Ts[x0 + i][y]; kv[i] = Ks[e0 + i][y]; }
            #pragma unroll
            for (int j = 0; j < 4; ++j)
                #pragma unroll
                for (int i = 0; i < 4; ++i) {
                    float2 p = cmulf(tv[i], kv[j]);
                    acc[j][i].x += p.x; acc[j][i].y += p.y;
                }
        }
    }
    #pragma unroll
    for (int j = 0; j < 4; ++j)
        #pragma unroll
        for (int i = 0; i < 4; ++i)
            xqkv[(size_t)bh * 4096 + (x0 + i) * 64 + (e0 + j)] = acc[j][i];
}

// bmm3: F[b,h,o,x] = sum_e xqkv[b,h,e,x] * W[h,e,o,x]. grid 512 = h*64 + x.
__global__ __launch_bounds__(256) void bmm3(
    const float2* __restrict__ xqkv, const float2* __restrict__ W,
    float2* __restrict__ F)
{
    const int h = blockIdx.x >> 6, x = blockIdx.x & 63;
    const int tid = threadIdx.x;
    __shared__ float2 Mw[64 * 64];   // [e][o]
    __shared__ float2 Vv[16 * 64];   // [b][e]
    for (int idx = tid; idx < 4096; idx += 256) {
        int e = idx >> 6, o = idx & 63;
        Mw[idx] = W[(size_t)(((h * 64 + e) * 64 + o)) * NTOPK + x];
    }
    for (int idx = tid; idx < 1024; idx += 256) {
        int b = idx >> 6, e = idx & 63;
        Vv[idx] = xqkv[(size_t)(b * 8 + h) * 4096 + x * 64 + e];
    }
    __syncthreads();
    const int o = tid & 63, bq = tid >> 6;
    float2 acc[4] = {};
    for (int e = 0; e < 64; ++e) {
        float2 m = Mw[e * 64 + o];
        #pragma unroll
        for (int j = 0; j < 4; ++j) {
            float2 p = cmulf(Vv[(bq + 4 * j) * 64 + e], m);
            acc[j].x += p.x; acc[j].y += p.y;
        }
    }
    #pragma unroll
    for (int j = 0; j < 4; ++j) {
        int b = bq + 4 * j;
        F[(size_t)((b * 8 + h) * 64 + o) * NTOPK + x] = acc[j];
    }
}

// --------------------------------------------------------------------------
// G operand for synthesis, scattered directly into A-fragment order planes.
__global__ void prepG(const float2* __restrict__ F, const int* __restrict__ qidx,
                      half_t* __restrict__ GH, half_t* __restrict__ GL)
{
    int gid = blockIdx.x * blockDim.x + threadIdx.x;   // 8192*64 exact
    int s2 = gid >> 6, x = gid & 63;
    int f = qidx[x];
    bool edge = (f == 0) || (f == NL / 2);
    float w = (edge ? 1.f : 2.f) * (1.f / (float)NL) * (1.f / 262144.f);
    float2 Fv = F[(size_t)s2 * NTOPK + x];
    h2 gr = splitf(Fv.x * w);
    h2 gi = splitf(edge ? 0.f : -Fv.y * w);
    int rtile = s2 >> 4, M = s2 & 15;
    #pragma unroll
    for (int r = 0; r < 2; ++r) {
        int K = 2 * x + r;
        int chunk = K >> 5, j = K & 31;
        size_t pos = (size_t)(rtile * 4 + chunk) * 512 + afragPos(M, j);
        GH[pos] = r ? gi.x : gr.x;
        GL[pos] = r ? gi.y : gr.y;
    }
}

// Basis2 planes: [t][k], k = 2x (cos) / 2x+1 (sin) at f = qidx[x].
__global__ void fillBasis2(const int* __restrict__ qidx,
                           half_t* __restrict__ BH, half_t* __restrict__ BL)
{
    int gid = blockIdx.x * blockDim.x + threadIdx.x;   // 2048*64 exact
    int t = gid >> 6, x = gid & 63;
    int f = qidx[x];
    int phase = (f * t) & (NL - 1);
    double th = (double)phase * (3.14159265358979323846 / 1024.0);
    h2 cc = splitf((float)cos(th));
    h2 ss = splitf((float)sin(th));
    BH[(size_t)t * 128 + 2 * x]     = cc.x;  BL[(size_t)t * 128 + 2 * x]     = cc.y;
    BH[(size_t)t * 128 + 2 * x + 1] = ss.x;  BL[(size_t)t * 128 + 2 * x + 1] = ss.y;
}

// Synthesis GEMM: out[8192 x 2048] = G * Basis2^T. grid (512,16), no LDS.
__global__ __launch_bounds__(256) void synth(
    const half_t* __restrict__ GH, const half_t* __restrict__ GL,
    const half_t* __restrict__ BH, const half_t* __restrict__ BL,
    float* __restrict__ out)
{
    const int rtile = blockIdx.x;
    const int tid = threadIdx.x;
    const int wave = tid >> 5, lane = tid & 31;
    const int lm = lane & 15, khalf = lane >> 4;
    const int ttile = blockIdx.y * 8 + wave;          // 0..127

    v8f acc = {};
    #pragma unroll
    for (int chunk = 0; chunk < 4; ++chunk) {
        size_t aofs = (size_t)(rtile * 4 + chunk) * 512 + lane * 16;
        size_t bofs = (size_t)(ttile * 16 + lm) * 128 + chunk * 32 + khalf * 16;
        v16h ah  = *(const v16h*)(GH + aofs);
        v16h al  = *(const v16h*)(GL + aofs);
        v16h bh_ = *(const v16h*)(BH + bofs);
        v16h bl_ = *(const v16h*)(BL + bofs);
        acc = __builtin_amdgcn_wmma_f32_16x16x32_f16(false, ah, false, bh_, (short)0, acc, false, false);
        acc = __builtin_amdgcn_wmma_f32_16x16x32_f16(false, al, false, bh_, (short)0, acc, false, false);
        acc = __builtin_amdgcn_wmma_f32_16x16x32_f16(false, ah, false, bl_, (short)0, acc, false, false);
    }
    #pragma unroll
    for (int r = 0; r < 8; ++r) {
        int row = rtile * 16 + r + (khalf ? 8 : 0);
        out[(size_t)row * NL + ttile * 16 + lm] = acc[r];
    }
}

// --------------------------------------------------------------------------
extern "C" void kernel_launch(void* const* d_in, const int* in_sizes, int n_in,
                              void* d_out, int out_size, void* d_ws, size_t ws_size,
                              hipStream_t stream)
{
    const float*  q  = (const float*)d_in[0];
    const float*  k  = (const float*)d_in[1];
    const float2* Wc = (const float2*)d_in[3];   // complex64 -> interleaved float2
    float* out = (float*)d_out;

    char* ws = (char*)d_ws;
    size_t off = 0;
    auto alloc = [&](size_t bytes) -> void* {
        void* p = (void*)(ws + off);
        off += (bytes + 255) & ~(size_t)255;
        return p;
    };
    float*  binsum = (float*)alloc((size_t)2 * NBP * 4);
    int*    idx    = (int*)alloc((size_t)2 * NTOPK * 4);
    half_t* CH     = (half_t*)alloc((size_t)NBP * NL * 2);
    half_t* CL     = (half_t*)alloc((size_t)NBP * NL * 2);
    half_t* SH     = (half_t*)alloc((size_t)NBP * NL * 2);
    half_t* SL     = (half_t*)alloc((size_t)NBP * NL * 2);
    half_t* AHq    = (half_t*)alloc((size_t)NSER * NL * 2);
    half_t* ALq    = (half_t*)alloc((size_t)NSER * NL * 2);
    half_t* AHk    = (half_t*)alloc((size_t)NSER * NL * 2);
    half_t* ALk    = (half_t*)alloc((size_t)NSER * NL * 2);
    float2* Xq     = (float2*)alloc((size_t)NSER * NTOPK * sizeof(float2));
    float2* Xk     = (float2*)alloc((size_t)NSER * NTOPK * sizeof(float2));
    float2* T      = (float2*)alloc((size_t)NBH * 4096 * sizeof(float2));
    float2* xqkv   = (float2*)alloc((size_t)NBH * 4096 * sizeof(float2));
    float2* F      = (float2*)alloc((size_t)NSER * NTOPK * sizeof(float2));
    half_t* GH     = (half_t*)alloc((size_t)NSER * 128 * 2);
    half_t* GL     = (half_t*)alloc((size_t)NSER * 128 * 2);
    half_t* BH     = (half_t*)alloc((size_t)NL * 128 * 2);
    half_t* BL     = (half_t*)alloc((size_t)NL * 128 * 2);

    initBins  <<<dim3((2 * NBP + 255) / 256), 256, 0, stream>>>(binsum);
    fillBasis1<<<dim3(NBP * NL / 256), 256, 0, stream>>>(CH, CL, SH, SL);
    splitQK   <<<dim3(512 * 64, 1, 2), 256, 0, stream>>>(q, k, AHq, ALq, AHk, ALk);
    pass1     <<<dim3(512, 9, 2), 256, 0, stream>>>(AHq, ALq, AHk, ALk, CH, CL, SH, SL, binsum);
    topk      <<<dim3(2), 256, 0, stream>>>(binsum, idx);
    pass2     <<<dim3(512, 1, 2), 128, 0, stream>>>(AHq, ALq, AHk, ALk, CH, CL, SH, SL, idx, Xq, Xk);
    bmm1      <<<dim3(NBH), 256, 0, stream>>>(Xq, Xk, T);
    bmm2      <<<dim3(NBH), 256, 0, stream>>>(T, Xk, xqkv);
    bmm3      <<<dim3(512), 256, 0, stream>>>(xqkv, Wc, F);
    prepG     <<<dim3(NSER * NTOPK / 256), 256, 0, stream>>>(F, idx, GH, GL);
    fillBasis2<<<dim3(NL * NTOPK / 256), 256, 0, stream>>>(idx, BH, BL);
    synth     <<<dim3(512, 16), 256, 0, stream>>>(GH, GL, BH, BL, out);
}
